// FullSNNDiffusionModel_35046933135762
// MI455X (gfx1250) — matrix operand order repaired and checked by reading, not compiled, not run on hardware
//
#include <hip/hip_runtime.h>
#include <hip/hip_bf16.h>
#include <math.h>
#include <stdint.h>

// ---------------- constants (match reference) ----------------
#define TT      4          // T_STEPS
#define BB      4          // batch
#define CC      64         // channels
#define C2      128        // 2*C
#define LL      4096       // length
#define NLAYERS 10
#define LT      32         // L-tile per block
#define XPITCH  80         // padded LDS row pitch, conv-input slabs (bf16 elems)
#define SPITCH  144        // padded LDS row pitch, spike buffer (bf16 elems)
#define ZPITCH  80         // padded LDS row pitch, z buffer (bf16 elems)
#define SZ_OFF  18432      // sZ elem offset inside smem (byte 36864, 32B aligned)

typedef __attribute__((ext_vector_type(8)))  __bf16 v8bf;
typedef __attribute__((ext_vector_type(16))) __bf16 v16bf;
typedef __attribute__((ext_vector_type(4)))  float  v4f;
typedef __attribute__((ext_vector_type(8)))  float  v8f;

// ---------------- bf16 <-> f32 (bit-exact RNE) ----------------
__device__ __forceinline__ __bf16 f2bf(float f) {
  unsigned u = __builtin_bit_cast(unsigned, f);
  unsigned r = (u + 0x7FFFu + ((u >> 16) & 1u)) >> 16;
  return __builtin_bit_cast(__bf16, (unsigned short)r);
}
__device__ __forceinline__ float bf2f(__bf16 h) {
  unsigned u = ((unsigned)__builtin_bit_cast(unsigned short, h)) << 16;
  return __builtin_bit_cast(float, u);
}

// A-matrix fragment, contiguous K order (matches B fragments):
// lane<16: K = 0..15; lane>=16: K = 16..31.
__device__ __forceinline__ v16bf load_afrag(const __bf16* row, int hi8) {
  v8bf lo = *(const v8bf*)(row + 2 * hi8);
  v8bf hi = *(const v8bf*)(row + 2 * hi8 + 8);
  return __builtin_shufflevector(lo, hi, 0,1,2,3,4,5,6,7,8,9,10,11,12,13,14,15);
}

#define WMMA_BF16(A, B, Cacc) \
  __builtin_amdgcn_wmma_f32_16x16x32_bf16(false, (A), false, (B), (short)0, (Cacc), false, false)

// ---- CDNA5 async global->LDS copy (16B per lane), ASYNCcnt-tracked ----
__device__ __forceinline__ void async_copy_b128(unsigned lds_off, const void* gsrc) {
  unsigned long long ga = (unsigned long long)(size_t)gsrc;
  asm volatile("global_load_async_to_lds_b128 %0, %1, off"
               :: "v"(lds_off), "v"(ga) : "memory");
}
__device__ __forceinline__ void wait_async0() {
#if __has_builtin(__builtin_amdgcn_s_wait_asynccnt)
  __builtin_amdgcn_s_wait_asynccnt(0);
#else
  asm volatile("s_wait_asynccnt 0" ::: "memory");
#endif
}

// ---------------- threefry2x32 (exact JAX semantics, key = [0,42]) ----------------
__device__ __forceinline__ void threefry2x32(unsigned k0, unsigned k1,
                                             unsigned x0, unsigned x1,
                                             unsigned* o0, unsigned* o1) {
  unsigned k2 = k0 ^ k1 ^ 0x1BD11BDAu;
  unsigned ks[3] = {k0, k1, k2};
  x0 += k0; x1 += k1;
  const int rotA[4] = {13, 15, 26, 6};
  const int rotB[4] = {17, 29, 16, 24};
  #pragma unroll
  for (int g = 1; g <= 5; ++g) {
    const int* R = (g & 1) ? rotA : rotB;
    #pragma unroll
    for (int r = 0; r < 4; ++r) {
      x0 += x1;
      x1 = (x1 << R[r]) | (x1 >> (32 - R[r]));
      x1 ^= x0;
    }
    x0 += ks[g % 3];
    x1 += ks[(g + 1) % 3] + (unsigned)g;
  }
  *o0 = x0; *o1 = x1;
}
__device__ __forceinline__ float bits2unif(unsigned b) {
  return __builtin_bit_cast(float, (b >> 9) | 0x3f800000u) - 1.0f;
}

// ======================================================================
// Kernel 1: convert/reorder weights to bf16 in workspace.
//  conv:  wsW[l][tap][co][ci] <- conv_w[l][co][ci][tap];  skip/res: copy
// ======================================================================
__global__ __launch_bounds__(256) void convert_weights_kernel(
    const float* __restrict__ conv_w, const float* __restrict__ skip_w,
    const float* __restrict__ res_w,
    __bf16* __restrict__ Wc, __bf16* __restrict__ Ws, __bf16* __restrict__ Wr) {
  const int NCONV = NLAYERS * 3 * C2 * CC;   // 245760
  const int NSR   = NLAYERS * CC * CC;       // 40960
  for (int idx = blockIdx.x * blockDim.x + threadIdx.x;
       idx < NCONV + 2 * NSR; idx += gridDim.x * blockDim.x) {
    if (idx < NCONV) {
      int ci  = idx & 63;
      int co  = (idx >> 6) & 127;
      int tap = (idx >> 13) % 3;
      int l   = idx / (3 * C2 * CC);
      Wc[idx] = f2bf(conv_w[((l * C2 + co) * CC + ci) * 3 + tap]);
    } else if (idx < NCONV + NSR) {
      int j = idx - NCONV;
      Ws[j] = f2bf(skip_w[j]);
    } else {
      int j = idx - NCONV - NSR;
      Wr[j] = f2bf(res_w[j]);
    }
  }
}

// ======================================================================
// Kernel 2: diffusion embedding -> proj[l][b][c], then fold proj through the
// conv taps:  ps[l][b][tap][co] = sum_ci conv_w[l][co][ci][tap] * proj[l][b][ci]
// (lets the layer kernel conv zero-padded x and add ps as a bias).
// ======================================================================
__global__ __launch_bounds__(256) void proj_kernel(
    const int* __restrict__ dstep,
    const float* __restrict__ w1, const float* __restrict__ b1,
    const float* __restrict__ w2, const float* __restrict__ b2,
    const float* __restrict__ dw, const float* __restrict__ db,
    const float* __restrict__ conv_w,
    float* __restrict__ projAll, float* __restrict__ psAll) {
  __shared__ float e1[BB][CC], e2[BB][CC];
  const int b = threadIdx.x >> 6;
  const int c = threadIdx.x & 63;
  const float d = (float)dstep[b];
  for (int l = 0; l < NLAYERS; ++l) {
    float t1 = d * w1[l * CC + c] + b1[l * CC + c];
    e1[b][c] = t1 / (1.0f + expf(-t1));            // silu
    __syncthreads();
    float a2 = b2[l * CC + c];
    for (int k = 0; k < CC; ++k) a2 += e1[b][k] * w2[(l * CC + c) * CC + k];
    e2[b][c] = a2;
    __syncthreads();
    float p = db[l * CC + c];
    for (int k = 0; k < CC; ++k) p += e2[b][k] * dw[(l * CC + c) * CC + k];
    projAll[(l * BB + b) * CC + c] = p;
    e1[b][c] = p;                                   // stash proj for ps stage
    __syncthreads();
    for (int it = 0; it < 6; ++it) {                // 1536 = BB*3*C2 outputs
      int idx = it * 256 + threadIdx.x;
      int co  = idx & 127;
      int tap = (idx >> 7) % 3;
      int bb  = idx / 384;
      float dot = 0.0f;
      for (int ci = 0; ci < CC; ++ci)
        dot += conv_w[(((size_t)l * C2 + co) * CC + ci) * 3 + tap] * e1[bb][ci];
      psAll[(((size_t)l * BB + bb) * 3 + tap) * C2 + co] = dot;
    }
    __syncthreads();
  }
}

// ======================================================================
// Kernel 3: Poisson encode (exact JAX threefry) + 1x1 input conv + LIF
//  -> X0[b][t][l][c] in {0,1}, stored bf16
// ======================================================================
__global__ __launch_bounds__(256) void encoder_kernel(
    const float* __restrict__ audio, const float* __restrict__ Win,
    const float* __restrict__ bin, __bf16* __restrict__ X0) {
  int idx = blockIdx.x * blockDim.x + threadIdx.x;   // 16384 = B*L
  int b = idx >> 12, l = idx & 4095;
  unsigned i0 = (unsigned)(b * LL + l);              // t=0 (partner t=2)
  unsigned i1 = i0 + 16384u;                         // t=1 (partner t=3)
  unsigned r0, r1, r2, r3;
  threefry2x32(0u, 42u, i0, i0 + 32768u, &r0, &r2);
  threefry2x32(0u, 42u, i1, i1 + 32768u, &r1, &r3);
  float a = audio[b * LL + l];
  float sp[TT];
  sp[0] = (bits2unif(r0) < a) ? 1.0f : 0.0f;
  sp[1] = (bits2unif(r1) < a) ? 1.0f : 0.0f;
  sp[2] = (bits2unif(r2) < a) ? 1.0f : 0.0f;
  sp[3] = (bits2unif(r3) < a) ? 1.0f : 0.0f;
  for (int c = 0; c < CC; ++c) {
    float wc = Win[c], bc = bin[c];
    float v = 0.0f;
    #pragma unroll
    for (int t = 0; t < TT; ++t) {
      float x = wc * sp[t] + bc;
      v += (x - v) / 1.2f;
      float s = (v - 0.5f >= 0.0f) ? 1.0f : 0.0f;
      v *= (1.0f - s);
      X0[((size_t)(b * TT + t) * LL + l) * CC + c] = f2bf(s);
    }
  }
}

// ======================================================================
// Kernel 4: fused residual layer (bf16 activations, bf16 WMMA).
//  async-copy x slabs (zero 'same' pad) -> conv C->2C -> +bias+ps -> LIF ->
//  z = sigmoid(gate)*tanh(filt) (affine on {0,1} spikes) ->
//  skip GEMM -> TS(+=) ;  res GEMM -> Xout = Xin + res.
// ======================================================================
__global__ __launch_bounds__(256) void layer_kernel(
    const __bf16* __restrict__ Xin, __bf16* __restrict__ Xout,
    float* __restrict__ TS,
    const __bf16* __restrict__ Wc, const __bf16* __restrict__ Ws,
    const __bf16* __restrict__ Wr, const float* __restrict__ psAll,
    const float* __restrict__ conv_b, const float* __restrict__ skip_b,
    const float* __restrict__ res_b, int layer, int dil, int first) {
  __shared__ __align__(64) __bf16 smem[3 * TT * LT * XPITCH];  // 61440 B
  __shared__ float sPs[3 * C2];
  __bf16* sX = smem;            // [3][T][LT][XPITCH]  conv-input slabs
  __bf16* sS = smem;            // [T][LT][SPITCH]     spikes   (bytes [0,36864))
  __bf16* sZ = smem + SZ_OFF;   // [T][LT][ZPITCH]     z values (bytes [36864,57344))

  const int tid = threadIdx.x;
  const int b   = blockIdx.y;
  const int l0  = blockIdx.x * LT;

  for (int it = tid; it < 3 * C2; it += 256)
    sPs[it] = psAll[((size_t)(layer * BB + b) * 3) * C2 + it];

  // ---- stage conv-input slabs: async 16B copies, zeros at 'same' padding ----
  for (int it = tid; it < 3 * TT * LT * 8; it += 256) {   // 16-byte chunks
    int cc  = it & 7;                 // 8 bf16 per chunk
    int ll  = (it >> 3) & (LT - 1);
    int t   = (it >> 8) & 3;
    int tap = it >> 10;
    unsigned dst = (unsigned)(size_t)(void*)
        (sX + ((size_t)((tap * TT + t) * LT + ll)) * XPITCH + cc * 8);
    int ls = l0 + ll + (tap - 1) * dil;
    if (ls < 0 || ls >= LL) {
      v8bf z;
      #pragma unroll
      for (int j = 0; j < 8; ++j) z[j] = f2bf(0.0f);
      *(v8bf*)(size_t)dst = z;
    } else {
      const __bf16* src = Xin + ((size_t)((b * TT + t) * LL + ls)) * CC + cc * 8;
      async_copy_b128(dst, src);
    }
  }
  wait_async0();
  __syncthreads();

  const int wv   = tid >> 5;
  const int lane = tid & 31;
  const int lrow = lane & 15;
  const int hi8  = (lane >> 4) << 3;    // A-frag chunk base
  const int hi16 = (lane >> 4) << 4;    // B-frag chunk base

  // ---- conv GEMM: M=128 (wave owns 16 rows), N=32 (2 tiles), K=3*64 ----
  const int Mb = wv * 16;
  v8f acc[TT][2] = {};
  #pragma unroll
  for (int tap = 0; tap < 3; ++tap) {
    #pragma unroll
    for (int kc = 0; kc < 2; ++kc) {
      const __bf16* arow =
          Wc + ((size_t)((layer * 3 + tap) * C2 + Mb + lrow)) * CC + kc * 32;
      v16bf A = load_afrag(arow, hi8);
      #pragma unroll
      for (int t = 0; t < TT; ++t) {
        #pragma unroll
        for (int nt = 0; nt < 2; ++nt) {
          const __bf16* brow =
              sX + ((size_t)((tap * TT + t) * LT + nt * 16 + lrow)) * XPITCH +
              kc * 32 + hi16;
          v16bf B = *(const v16bf*)brow;
          acc[t][nt] = WMMA_BF16(A, B, acc[t][nt]);
        }
      }
    }
  }
  __syncthreads();   // all waves done reading sX (sS aliases it)

  // ---- +bias (+ per-boundary proj-fold), LIF over T, spikes -> sS ----
  {
    const float* cb = conv_b + layer * C2 + Mb + hi8;
    float b1v[8], p0[8], p2[8];
    #pragma unroll
    for (int e = 0; e < 8; ++e) {
      b1v[e] = cb[e] + sPs[C2 + Mb + hi8 + e];     // center tap always in-range
      p0[e]  = sPs[Mb + hi8 + e];
      p2[e]  = sPs[2 * C2 + Mb + hi8 + e];
    }
    #pragma unroll
    for (int nt = 0; nt < 2; ++nt) {
      int ll = nt * 16 + lrow;
      int l  = l0 + ll;
      float f0 = (l >= dil) ? 1.0f : 0.0f;
      float f2 = (l + dil < LL) ? 1.0f : 0.0f;
      float vm[8];
      #pragma unroll
      for (int e = 0; e < 8; ++e) vm[e] = 0.0f;
      #pragma unroll
      for (int t = 0; t < TT; ++t) {
        v8bf sv;
        #pragma unroll
        for (int e = 0; e < 8; ++e) {
          float x = acc[t][nt][e] + b1v[e] + f0 * p0[e] + f2 * p2[e];
          vm[e] += (x - vm[e]) / 1.2f;
          float s = (vm[e] - 0.5f >= 0.0f) ? 1.0f : 0.0f;
          vm[e] *= (1.0f - s);
          sv[e] = f2bf(s);
        }
        *(v8bf*)(sS + ((size_t)(t * LT + ll)) * SPITCH + Mb + hi8) = sv;
      }
    }
  }
  __syncthreads();

  // ---- z = sigmoid(gate)*tanh(filt), materialized once into sZ ----
  // spikes are exactly {0,1}: sigmoid(g)=0.5+0.23105858*g, tanh(f)=0.76159416*f
  for (int it = tid; it < TT * LT * 4; it += 256) {   // 16-channel chunks
    int c16 = (it & 3) << 4;
    int ll  = (it >> 2) & (LT - 1);
    int t   = it >> 7;
    const __bf16* row = sS + ((size_t)(t * LT + ll)) * SPITCH + c16;
    v16bf gv = *(const v16bf*)row;
    v16bf fv = *(const v16bf*)(row + CC);
    v16bf Z;
    #pragma unroll
    for (int j = 0; j < 16; ++j)
      Z[j] = f2bf((0.5f + 0.23105857863f * bf2f(gv[j])) *
                  (0.76159415595f * bf2f(fv[j])));
    *(v16bf*)(sZ + ((size_t)(t * LT + ll)) * ZPITCH + c16) = Z;
  }
  __syncthreads();

  // ---- skip/res GEMMs: waves 0-3 -> skip Mtile, waves 4-7 -> res Mtile ----
  const bool isSkip = (wv < 4);
  const int  Mb2    = (wv & 3) * 16;
  const __bf16* W2 =
      (isSkip ? Ws : Wr) + ((size_t)(layer * CC + Mb2 + lrow)) * CC;
  v8f acc2[TT][2] = {};
  #pragma unroll
  for (int kc = 0; kc < 2; ++kc) {
    v16bf A = load_afrag(W2 + kc * 32, hi8);
    #pragma unroll
    for (int t = 0; t < TT; ++t) {
      #pragma unroll
      for (int nt = 0; nt < 2; ++nt) {
        int ll = nt * 16 + lrow;
        v16bf Z = *(const v16bf*)(sZ + ((size_t)(t * LT + ll)) * ZPITCH +
                                  kc * 32 + hi16);
        acc2[t][nt] = WMMA_BF16(A, Z, acc2[t][nt]);
      }
    }
  }

  // ---- epilogue: TS += skip (fp32) ; Xout = Xin + res (bf16) ----
  {
    const float* bp = (isSkip ? skip_b : res_b) + layer * CC + Mb2 + hi8;
    v4f b20 = *(const v4f*)bp;
    v4f b21 = *(const v4f*)(bp + 4);
    #pragma unroll
    for (int t = 0; t < TT; ++t) {
      #pragma unroll
      for (int nt = 0; nt < 2; ++nt) {
        int ll = nt * 16 + lrow;
        size_t off = ((size_t)((b * TT + t) * LL + l0 + ll)) * CC + Mb2 + hi8;
        if (isSkip) {
          v4f lo, hi;
          #pragma unroll
          for (int e = 0; e < 4; ++e) {
            lo[e] = acc2[t][nt][e] + b20[e];
            hi[e] = acc2[t][nt][4 + e] + b21[e];
          }
          float* dst = TS + off;
          if (!first) {
            lo += *(const v4f*)dst;
            hi += *(const v4f*)(dst + 4);
          }
          *(v4f*)dst = lo;
          *(v4f*)(dst + 4) = hi;
        } else {
          v8bf xi = *(const v8bf*)(Xin + off);
          v8bf xo;
          #pragma unroll
          for (int e = 0; e < 8; ++e) {
            float bias = (e < 4) ? b20[e] : b21[e - 4];
            xo[e] = f2bf(bf2f(xi[e]) + acc2[t][nt][e] + bias);
          }
          *(v8bf*)(Xout + off) = xo;
        }
      }
    }
  }
}

// ======================================================================
// Kernel 5: relu(total_skip) -> 1x1 conv C->1 -> LIF -> sum over T
// ======================================================================
__global__ __launch_bounds__(256) void output_kernel(
    const float* __restrict__ TS, const float* __restrict__ Wout,
    const float* __restrict__ bout, float* __restrict__ out) {
  int idx = blockIdx.x * blockDim.x + threadIdx.x;   // B*L
  int b = idx >> 12, l = idx & 4095;
  float v = 0.0f, ssum = 0.0f;
  #pragma unroll
  for (int t = 0; t < TT; ++t) {
    const float* p = TS + ((size_t)((b * TT + t) * LL + l)) * CC;
    float dot = 0.0f;
    for (int c = 0; c < CC; c += 4) {
      v4f x = *(const v4f*)(p + c);
      #pragma unroll
      for (int e = 0; e < 4; ++e) dot += fmaxf(x[e], 0.0f) * Wout[c + e];
    }
    dot += bout[0];
    v += (dot - v) / 1.2f;
    float s = (v - 0.5f >= 0.0f) ? 1.0f : 0.0f;
    v *= (1.0f - s);
    ssum += s;
  }
  out[b * LL + l] = ssum;
}

// ======================================================================
// Host launcher
// ======================================================================
extern "C" void kernel_launch(void* const* d_in, const int* in_sizes, int n_in,
                              void* d_out, int out_size, void* d_ws, size_t ws_size,
                              hipStream_t stream) {
  (void)in_sizes; (void)n_in; (void)out_size; (void)ws_size;

  const float* audio   = (const float*)d_in[0];
  const int*   dstep   = (const int*)d_in[1];
  const float* W_in    = (const float*)d_in[2];
  const float* b_in    = (const float*)d_in[3];
  const float* demb_w1 = (const float*)d_in[4];
  const float* demb_b1 = (const float*)d_in[5];
  const float* demb_w2 = (const float*)d_in[6];
  const float* demb_b2 = (const float*)d_in[7];
  const float* dproj_w = (const float*)d_in[8];
  const float* dproj_b = (const float*)d_in[9];
  const float* conv_w  = (const float*)d_in[10];
  const float* conv_b  = (const float*)d_in[11];
  const float* skip_w  = (const float*)d_in[12];
  const float* skip_b  = (const float*)d_in[13];
  const float* res_w   = (const float*)d_in[14];
  const float* res_b   = (const float*)d_in[15];
  const float* W_out   = (const float*)d_in[16];
  const float* b_out   = (const float*)d_in[17];
  float* out = (float*)d_out;

  // workspace layout (all regions 256B aligned)
  const size_t actBF = (size_t)BB * TT * LL * CC * 2;   // bf16 activations, 8 MiB
  const size_t actF  = (size_t)BB * TT * LL * CC * 4;   // fp32 skip acc, 16 MiB
  char* ws = (char*)d_ws;
  __bf16* X0 = (__bf16*)(ws);
  __bf16* X1 = (__bf16*)(ws + actBF);
  float*  TS = (float*)(ws + 2 * actBF);
  size_t o = 2 * actBF + actF;
  __bf16* Wc = (__bf16*)(ws + o); o += (size_t)NLAYERS * 3 * C2 * CC * 2; o = (o + 255) & ~(size_t)255;
  __bf16* Ws = (__bf16*)(ws + o); o += (size_t)NLAYERS * CC * CC * 2;     o = (o + 255) & ~(size_t)255;
  __bf16* Wr = (__bf16*)(ws + o); o += (size_t)NLAYERS * CC * CC * 2;     o = (o + 255) & ~(size_t)255;
  float* projAll = (float*)(ws + o); o += (size_t)NLAYERS * BB * CC * 4;  o = (o + 255) & ~(size_t)255;
  float* psAll   = (float*)(ws + o);

  convert_weights_kernel<<<512, 256, 0, stream>>>(conv_w, skip_w, res_w, Wc, Ws, Wr);
  proj_kernel<<<1, 256, 0, stream>>>(dstep, demb_w1, demb_b1, demb_w2, demb_b2,
                                     dproj_w, dproj_b, conv_w, projAll, psAll);
  encoder_kernel<<<(BB * LL) / 256, 256, 0, stream>>>(audio, W_in, b_in, X0);

  __bf16* xbuf[2] = {X0, X1};
  for (int i = 0; i < NLAYERS; ++i) {
    int dil = 1 << (i % 10);
    dim3 grid(LL / LT, BB);
    layer_kernel<<<grid, 256, 0, stream>>>(
        xbuf[i & 1], xbuf[(i + 1) & 1], TS, Wc, Ws, Wr, psAll,
        conv_b, skip_b, res_b, i, dil, (i == 0) ? 1 : 0);
  }

  output_kernel<<<(BB * LL) / 256, 256, 0, stream>>>(TS, W_out, b_out, out);
}
